// MultiHeadedAttention_77369540870789
// MI455X (gfx1250) — compile-verified, matrix operand order checked by reading, not run on hardware
//
#include <hip/hip_runtime.h>
#include <stdint.h>

// MI455X / gfx1250 — wave32, WMMA f16 (16x16x32, f32 accumulate),
// async global->LDS staging (ASYNCcnt) for the projection A-tiles.
typedef __attribute__((ext_vector_type(16))) _Float16 v16h;
typedef __attribute__((ext_vector_type(8)))  float    v8f;

#define BB 4
#define SS 1024
#define DD 1024
#define HH 16
#define HDIM 64
#define NEGV (-1000000000.0f)

// ---------------------------------------------------------------------------
// Kernel 1: fused QKV projection  Y = X @ W^T + b
//   - block = 8 waves, all sharing one 16-row A tile of X
//   - A tile streamed global->LDS with async b128 copies, double buffered
//   - each wave computes a 16x32 output tile (2 WMMA accumulators, shared A)
//   - q,k written f16 head-major (B,H,S,HD); v written TRANSPOSED (B,H,HD,S)
//     so the PV GEMM's B-operand becomes contiguous.
// ---------------------------------------------------------------------------
__global__ __launch_bounds__(256) void qkv_proj_kernel(
    const float* __restrict__ Xq, const float* __restrict__ Xk, const float* __restrict__ Xv,
    const float* __restrict__ Wq, const float* __restrict__ bq,
    const float* __restrict__ Wk, const float* __restrict__ bk,
    const float* __restrict__ Wv, const float* __restrict__ bv,
    _Float16* __restrict__ qf, _Float16* __restrict__ kf, _Float16* __restrict__ vf)
{
    // grid: 3 projs * 256 m-tiles * 4 n-blocks (each n-block = 16 col tiles)
    __shared__ __align__(16) float xbuf[2][16][64];   // 8 KB, double buffered

    int tid  = threadIdx.x;
    int wv   = tid >> 5;
    int lane = tid & 31;
    int half = lane >> 4;
    int lcol = lane & 15;

    int proj  = blockIdx.x >> 10;          // /1024
    int rem   = blockIdx.x & 1023;
    int mt    = rem >> 2;
    int ntblk = rem & 3;

    const float* X; const float* W; const float* bias; _Float16* dst;
    if (proj == 0)      { X = Xq; W = Wq; bias = bq; dst = qf; }
    else if (proj == 1) { X = Xk; W = Wk; bias = bk; dst = kf; }
    else                { X = Xv; W = Wv; bias = bv; dst = vf; }

    int row0 = mt * 16;
    int colA = (ntblk * 16 + wv * 2) * 16 + lcol;   // first output column
    int colB = colA + 16;                           // second output column

    // async stage of one 16x64-float chunk of X: each thread copies 16 bytes
    int srow = tid >> 4;        // 0..15
    int sseg = tid & 15;        // 0..15 (64 floats -> 4 per seg)
    const float* xrow = X + (size_t)(row0 + srow) * DD + sseg * 4;

    auto issue_chunk = [&](int c) {
        const float*   g    = xrow + c * 64;
        unsigned       loff = (unsigned)(uintptr_t)&xbuf[c & 1][srow][sseg * 4];
        asm volatile("global_load_async_to_lds_b128 %0, %1, off"
                     :: "v"(loff), "v"(g) : "memory");
    };

    issue_chunk(0);

    v8f acc0 = {}, acc1 = {};
    for (int c = 0; c < 16; ++c) {                 // 16 chunks of 64 K-values
        if (c + 1 < 16) {
            issue_chunk(c + 1);
            asm volatile("s_wait_asynccnt 0x1" ::: "memory");
        } else {
            asm volatile("s_wait_asynccnt 0x0" ::: "memory");
        }
        __syncthreads();

        #pragma unroll
        for (int kk = 0; kk < 2; ++kk) {           // two 32-deep WMMA steps
            int kloc  = kk * 32 + half * 16;       // within chunk
            int kglob = c * 64 + kloc;

            // A fragment from LDS (convert f32 -> f16)
            const float4* xp = reinterpret_cast<const float4*>(&xbuf[c & 1][lcol][kloc]);
            v16h a;
            #pragma unroll
            for (int u = 0; u < 4; ++u) {
                float4 f = xp[u];
                a[4*u+0] = (_Float16)f.x; a[4*u+1] = (_Float16)f.y;
                a[4*u+2] = (_Float16)f.z; a[4*u+3] = (_Float16)f.w;
            }
            // B fragments from W (rows = output columns), contiguous
            const float4* bpA = reinterpret_cast<const float4*>(W + (size_t)colA * DD + kglob);
            const float4* bpB = reinterpret_cast<const float4*>(W + (size_t)colB * DD + kglob);
            v16h b0, b1;
            #pragma unroll
            for (int u = 0; u < 4; ++u) {
                float4 fa = bpA[u]; float4 fb = bpB[u];
                b0[4*u+0] = (_Float16)fa.x; b0[4*u+1] = (_Float16)fa.y;
                b0[4*u+2] = (_Float16)fa.z; b0[4*u+3] = (_Float16)fa.w;
                b1[4*u+0] = (_Float16)fb.x; b1[4*u+1] = (_Float16)fb.y;
                b1[4*u+2] = (_Float16)fb.z; b1[4*u+3] = (_Float16)fb.w;
            }
            acc0 = __builtin_amdgcn_wmma_f32_16x16x32_f16(false, a, false, b0,
                                                          (short)0, acc0, false, false);
            acc1 = __builtin_amdgcn_wmma_f32_16x16x32_f16(false, a, false, b1,
                                                          (short)0, acc1, false, false);
        }
        __syncthreads();
    }

    // bias + store (v goes out transposed: (B,H,HD,S))
    auto store_tile = [&](int col, const v8f& acc, float bvv) {
        int hh = col >> 6;
        int hd = col & 63;
        #pragma unroll
        for (int r = 0; r < 8; ++r) {
            int m  = row0 + half * 8 + r;
            int bi = m >> 10;
            int s  = m & 1023;
            size_t idx = (proj == 2)
                ? (((size_t)(bi * HH + hh)) * HDIM + hd) * SS + s
                : (((size_t)(bi * HH + hh)) * SS + s) * HDIM + hd;
            dst[idx] = (_Float16)(acc[r] + bvv);
        }
    };
    store_tile(colA, acc0, bias[colA]);
    store_tile(colB, acc1, bias[colB]);
}

// ---------------------------------------------------------------------------
// Kernel 2: fused attention per (b, h, 16-row tile):
//   scores (WMMA) -> 3-way softmax (prob/time/rel) -> mix -> prob (f32 out,
//   f16 LDS) -> prob @ V (WMMA, V pre-transposed, 8-wave split-K) -> out
// ---------------------------------------------------------------------------
__global__ __launch_bounds__(256) void attention_kernel(
    const _Float16* __restrict__ qf, const _Float16* __restrict__ kf,
    const _Float16* __restrict__ vT,
    const float* __restrict__ rel, const float* __restrict__ timestamp,
    const float* __restrict__ l1p, const float* __restrict__ l2p,
    float* __restrict__ outp, float* __restrict__ probp)
{
    __shared__ __align__(32) _Float16 sc[16 * SS];   // 32 KB: logits then prob(f16)
    __shared__ float red[3][16][16];
    __shared__ float rmax[3][16];
    __shared__ float rsum[3][16];
    __shared__ float pacc[4][16][16];                // split-K partials for PV
    __shared__ float l12[2];

    int tid  = threadIdx.x;
    int wv   = tid >> 5;
    int lane = tid & 31;
    int half = lane >> 4;
    int lcol = lane & 15;

    int bh   = blockIdx.x >> 6;          // b*H + h
    int rt   = blockIdx.x & 63;
    int b    = bh >> 4;
    int h    = bh & 15;
    int row0 = rt * 16;

    if (tid == 0) { l12[0] = l1p[0]; l12[1] = l2p[0]; }

    const _Float16* qbase = qf + (size_t)bh * SS * HDIM;
    const _Float16* kbase = kf + (size_t)bh * SS * HDIM;

    // ---- Phase 1: scores = q k^T / 8 via WMMA ----
    v16h aq0 = *reinterpret_cast<const v16h*>(qbase + (size_t)(row0 + lcol) * HDIM + half * 16);
    v16h aq1 = *reinterpret_cast<const v16h*>(qbase + (size_t)(row0 + lcol) * HDIM + 32 + half * 16);

    for (int ct = wv; ct < SS / 16; ct += 8) {
        v8f acc = {};
        v16h b0 = *reinterpret_cast<const v16h*>(kbase + (size_t)(ct * 16 + lcol) * HDIM + half * 16);
        acc = __builtin_amdgcn_wmma_f32_16x16x32_f16(false, aq0, false, b0, (short)0, acc, false, false);
        v16h b1 = *reinterpret_cast<const v16h*>(kbase + (size_t)(ct * 16 + lcol) * HDIM + 32 + half * 16);
        acc = __builtin_amdgcn_wmma_f32_16x16x32_f16(false, aq1, false, b1, (short)0, acc, false, false);
        #pragma unroll
        for (int r = 0; r < 8; ++r)
            sc[(half * 8 + r) * SS + ct * 16 + lcol] = (_Float16)(acc[r] * 0.125f);
    }
    __syncthreads();

    // ---- Phase 2: per-row max & sum for the three softmaxes ----
    int rowi  = tid >> 4;
    int seg   = tid & 15;
    int row_g = row0 + rowi;
    const float* ts_row  = timestamp + ((size_t)b * SS + row_g) * SS;
    const float* rel_row = rel       + ((size_t)b * SS + row_g) * SS;

    float pm0 = -3.0e38f, pm1 = -3.0e38f, pm2 = -3.0e38f;
    for (int c = seg * 64; c < seg * 64 + 64; ++c) {
        bool keep = (c <= row_g);
        float sv = keep ? (float)sc[rowi * SS + c] : NEGV;
        float tv = keep ? __expf(-fabsf(ts_row[c])) : NEGV;
        float rm = keep ? rel_row[c] : 0.0f;
        float rv = (rm == 0.0f) ? -100000.0f : rm;
        pm0 = fmaxf(pm0, sv); pm1 = fmaxf(pm1, tv); pm2 = fmaxf(pm2, rv);
    }
    red[0][rowi][seg] = pm0; red[1][rowi][seg] = pm1; red[2][rowi][seg] = pm2;
    __syncthreads();
    if (tid < 48) {
        int f = tid >> 4, rr = tid & 15;
        float m = -3.0e38f;
        for (int s = 0; s < 16; ++s) m = fmaxf(m, red[f][rr][s]);
        rmax[f][rr] = m;
    }
    __syncthreads();

    float m0 = rmax[0][rowi], m1 = rmax[1][rowi], m2 = rmax[2][rowi];
    float s0 = 0.f, s1 = 0.f, s2 = 0.f;
    for (int c = seg * 64; c < seg * 64 + 64; ++c) {
        bool keep = (c <= row_g);
        float sv = keep ? (float)sc[rowi * SS + c] : NEGV;
        float tv = keep ? __expf(-fabsf(ts_row[c])) : NEGV;
        float rm = keep ? rel_row[c] : 0.0f;
        float rv = (rm == 0.0f) ? -100000.0f : rm;
        s0 += __expf(sv - m0); s1 += __expf(tv - m1); s2 += __expf(rv - m2);
    }
    red[0][rowi][seg] = s0; red[1][rowi][seg] = s1; red[2][rowi][seg] = s2;
    __syncthreads();
    if (tid < 48) {
        int f = tid >> 4, rr = tid & 15;
        float s = 0.f;
        for (int q = 0; q < 16; ++q) s += red[f][rr][q];
        rsum[f][rr] = s;
    }
    __syncthreads();

    // ---- Phase 3: mix probabilities, write f32 prob, stash f16 prob in LDS ----
    float is0 = 1.0f / rsum[0][rowi], is1 = 1.0f / rsum[1][rowi], is2 = 1.0f / rsum[2][rowi];
    float L1 = l12[0], L2 = l12[1];
    float* prob_row = probp + ((size_t)bh * SS + row_g) * SS;
    for (int c = seg * 64; c < seg * 64 + 64; ++c) {
        bool keep = (c <= row_g);
        float sv = keep ? (float)sc[rowi * SS + c] : NEGV;
        float tv = keep ? __expf(-fabsf(ts_row[c])) : NEGV;
        float rm = keep ? rel_row[c] : 0.0f;
        float rv = (rm == 0.0f) ? -100000.0f : rm;
        float pa = __expf(sv - m0) * is0;
        float ta = __expf(tv - m1) * is1;
        float ra = __expf(rv - m2) * is2;
        float p  = (1.0f - L2) * pa + L2 * ta;
        p = (1.0f - L1) * p + L1 * ra;
        prob_row[c] = p;
        sc[rowi * SS + c] = (_Float16)p;   // reuse LDS as f16 prob for PV GEMM
    }
    __syncthreads();

    // ---- Phase 4: out = prob @ V via WMMA, all 8 waves: 4 hd-tiles x 2 K-halves.
    // V is pre-transposed (B,H,HD,S) so the B fragment is 16 contiguous halves.
    {
        int tile  = wv & 3;
        int khalf = wv >> 2;
        int hd    = tile * 16 + lcol;
        const _Float16* vrow = vT + ((size_t)bh * HDIM + hd) * SS;

        v8f acc = {};
        int kbeg = khalf * (SS / 2);
        for (int k0 = kbeg; k0 < kbeg + SS / 2; k0 += 32) {
            int kb = k0 + half * 16;
            v16h a  = *reinterpret_cast<const v16h*>(&sc[lcol * SS + kb]);
            v16h bf = *reinterpret_cast<const v16h*>(vrow + kb);
            acc = __builtin_amdgcn_wmma_f32_16x16x32_f16(false, a, false, bf,
                                                         (short)0, acc, false, false);
        }
        if (wv >= 4) {
            #pragma unroll
            for (int r = 0; r < 8; ++r)
                pacc[tile][half * 8 + r][lcol] = acc[r];
        }
        __syncthreads();
        if (wv < 4) {
            #pragma unroll
            for (int r = 0; r < 8; ++r) {
                int sg = row0 + half * 8 + r;
                float v = acc[r] + pacc[tile][half * 8 + r][lcol];
                outp[((size_t)(b * SS + sg)) * DD + h * HDIM + hd] = v;
            }
        }
    }
}

// ---------------------------------------------------------------------------
// Host-side launcher
// ---------------------------------------------------------------------------
extern "C" void kernel_launch(void* const* d_in, const int* in_sizes, int n_in,
                              void* d_out, int out_size, void* d_ws, size_t ws_size,
                              hipStream_t stream) {
    (void)in_sizes; (void)n_in; (void)out_size; (void)ws_size;

    const float* query     = (const float*)d_in[0];
    const float* key       = (const float*)d_in[1];
    const float* value     = (const float*)d_in[2];
    const float* rel       = (const float*)d_in[3];
    const float* l1        = (const float*)d_in[4];
    const float* l2        = (const float*)d_in[5];
    const float* timestamp = (const float*)d_in[6];
    // d_in[7] encode_pos, d_in[8..9] pos embeds, d_in[10] mask: unused
    // (encode_pos==0 path; causal mask computed analytically)
    const float* Wq = (const float*)d_in[11];
    const float* bq = (const float*)d_in[12];
    const float* Wk = (const float*)d_in[13];
    const float* bk = (const float*)d_in[14];
    const float* Wv = (const float*)d_in[15];
    const float* bv = (const float*)d_in[16];

    // Workspace: f16 q/k in (B,H,S,HD), v transposed in (B,H,HD,S); 24 MB total
    _Float16* qf = (_Float16*)d_ws;
    _Float16* kf = qf + (size_t)BB * HH * SS * HDIM;
    _Float16* vf = kf + (size_t)BB * HH * SS * HDIM;

    float* outp  = (float*)d_out;                       // (B,S,D) = 4M floats
    float* probp = outp + (size_t)BB * SS * DD;         // (B,H,S,S) = 64M floats

    // 3 projs * 256 m-tiles * 4 n-blocks
    qkv_proj_kernel<<<3 * 256 * 4, 256, 0, stream>>>(
        query, key, value, Wq, bq, Wk, bk, Wv, bv, qf, kf, vf);

    // one block per (b, h, 16-row tile): 4*16*64 = 4096 blocks
    attention_kernel<<<BB * HH * (SS / 16), 256, 0, stream>>>(
        qf, kf, vf, rel, timestamp, l1, l2, outp, probp);
}
